// FloatEmbedding_78615081386413
// MI455X (gfx1250) — matrix operand order repaired and checked by reading, not compiled, not run on hardware
//
#include <hip/hip_runtime.h>

typedef __attribute__((ext_vector_type(16))) _Float16 v16h;
typedef __attribute__((ext_vector_type(8)))  float    v8f;

namespace {

constexpr int KF = 256;              // feature dim (K)
constexpr int NG = 256;              // output dim (N)
constexpr int STRIPES = 2;           // 16-row M stripes per wave
constexpr int ROWS_PER_WAVE = 16 * STRIPES;            // 32
constexpr int WAVES = 2;             // 64-thread block
constexpr int ROWS_PER_BLOCK = ROWS_PER_WAVE * WAVES;  // 64
constexpr int NT = NG / 16;          // 16 N-tiles
constexpr int KS = KF / 32;          // 8 K-steps per tile
// packed W: [nt][s][lane][8 x uint32 (16 f16)] = 32768 uint32 = 128 KiB
constexpr size_t PACKED_WORDS = (size_t)NT * KS * 32 * 8;
constexpr size_t PACKED_BYTES = PACKED_WORDS * 4;

__device__ __forceinline__ uint32_t rotl32(uint32_t x, int n) {
  return (x << n) | (x >> (32 - n));
}

// Exact threefry-2x32 (20 rounds), matching jax.random's generator.
__device__ __forceinline__ void tf2x32(uint32_t k0, uint32_t k1,
                                       uint32_t x0, uint32_t x1,
                                       uint32_t &o0, uint32_t &o1) {
  const uint32_t k2 = k0 ^ k1 ^ 0x1BD11BDAu;
  x0 += k0; x1 += k1;
  x0 += x1; x1 = rotl32(x1, 13); x1 ^= x0;
  x0 += x1; x1 = rotl32(x1, 15); x1 ^= x0;
  x0 += x1; x1 = rotl32(x1, 26); x1 ^= x0;
  x0 += x1; x1 = rotl32(x1,  6); x1 ^= x0;
  x0 += k1; x1 += k2 + 1u;
  x0 += x1; x1 = rotl32(x1, 17); x1 ^= x0;
  x0 += x1; x1 = rotl32(x1, 29); x1 ^= x0;
  x0 += x1; x1 = rotl32(x1, 16); x1 ^= x0;
  x0 += x1; x1 = rotl32(x1, 24); x1 ^= x0;
  x0 += k2; x1 += k0 + 2u;
  x0 += x1; x1 = rotl32(x1, 13); x1 ^= x0;
  x0 += x1; x1 = rotl32(x1, 15); x1 ^= x0;
  x0 += x1; x1 = rotl32(x1, 26); x1 ^= x0;
  x0 += x1; x1 = rotl32(x1,  6); x1 ^= x0;
  x0 += k0; x1 += k1 + 3u;
  x0 += x1; x1 = rotl32(x1, 17); x1 ^= x0;
  x0 += x1; x1 = rotl32(x1, 29); x1 ^= x0;
  x0 += x1; x1 = rotl32(x1, 16); x1 ^= x0;
  x0 += x1; x1 = rotl32(x1, 24); x1 ^= x0;
  x0 += k1; x1 += k2 + 4u;
  x0 += x1; x1 = rotl32(x1, 13); x1 ^= x0;
  x0 += x1; x1 = rotl32(x1, 15); x1 ^= x0;
  x0 += x1; x1 = rotl32(x1, 26); x1 ^= x0;
  x0 += x1; x1 = rotl32(x1,  6); x1 ^= x0;
  o0 = x0 + k2; o1 = x1 + k0 + 5u;
}

// jax.random.uniform bit trick: [1,2) mantissa fill, minus 1, times 100.
__device__ __forceinline__ float u01x100(uint32_t bits) {
  uint32_t f = (bits >> 9) | 0x3f800000u;
  return (__uint_as_float(f) - 1.0f) * 100.0f;
}

} // namespace

// One-shot: convert W (f32, [g][f] row-major) into f16 B-fragment order.
// t = ((nt*KS + s)*32 + lane)*8 + v  ->  pw[t] packs W[g][f], W[g][f+1]
// with g = nt*16 + lane%16, f = 32*s + 16*(lane/16) + 2*v.
__global__ __launch_bounds__(256)
void pack_W_f16(const float* __restrict__ W, uint32_t* __restrict__ pw) {
  const int t    = blockIdx.x * blockDim.x + threadIdx.x;  // 0..32767
  const int v    = t & 7;
  const int lane = (t >> 3) & 31;
  const int s    = (t >> 8) & 7;
  const int nt   = t >> 11;
  const int gcol = nt * 16 + (lane & 15);
  const int f    = 32 * s + 16 * (lane >> 4) + 2 * v;
  const float* wr = W + (size_t)gcol * KF + f;
  union { uint32_t u; _Float16 h[2]; } p;
  p.h[0] = (_Float16)wr[0];
  p.h[1] = (_Float16)wr[1];
  pw[t] = p.u;
}

template <bool PACKED>
__global__ __launch_bounds__(WAVES * 32)
void float_embed_linear_wmma(const float* __restrict__ x,
                             const float* __restrict__ W,
                             const uint32_t* __restrict__ pw,
                             const float* __restrict__ bias,
                             float* __restrict__ out) {
  // f16 emb staging: 2 waves x 32 rows x 256 features = 32 KiB LDS
  __shared__ _Float16 s_emb[WAVES][ROWS_PER_WAVE][KF];

  const int tid  = threadIdx.x;
  const int wave = tid >> 5;
  const int lane = tid & 31;
  const int mrow = lane & 15;   // row within a 16-row stripe
  const int half = lane >> 4;   // lane group 0..15 vs 16..31

  const int row0 = blockIdx.x * ROWS_PER_BLOCK + wave * ROWS_PER_WAVE;

  // ---- RNG: fold_in(key(42), bitcast(x)), 64 counter pairs per lane/stripe --
  const uint32_t cbase = (uint32_t)(64 * half);
  #pragma unroll
  for (int t = 0; t < STRIPES; ++t) {
    uint32_t fk0, fk1;
    uint32_t seed = __float_as_uint(x[row0 + t * 16 + mrow]);
    tf2x32(0u, 42u, 0u, seed, fk0, fk1);   // folded key
    _Float16* erow = &s_emb[wave][t * 16 + mrow][0];
    #pragma unroll 4
    for (uint32_t j = 0; j < 64; ++j) {
      uint32_t c = cbase + j, o0, o1;
      tf2x32(fk0, fk1, c, c + 128u, o0, o1);  // iota split (c, c+128)
      erow[c]        = (_Float16)u01x100(o0);
      erow[c + 128u] = (_Float16)u01x100(o1);
    }
  }
  __syncthreads();

  // ---- A fragments: 16x32 f16 per K-step, contiguous k order (k = 16*half + elem) ----
  // lane<16: V0-7 -> K=2v..2v+1; lanes>=16: +16
  union AU { v16h v; uint32_t u[8]; };
  AU afrag[STRIPES][KS];
  #pragma unroll
  for (int t = 0; t < STRIPES; ++t) {
    const uint32_t* rowp = (const uint32_t*)&s_emb[wave][t * 16 + mrow][0];
    #pragma unroll
    for (int s = 0; s < KS; ++s) {
      #pragma unroll
      for (int v = 0; v < 8; ++v) {
        int k = 32 * s + 2 * v + 16 * half;  // even
        afrag[t][s].u[v] = rowp[k >> 1];
      }
    }
  }

  // ---- N-tile loop: D = A * W^T + b via v_wmma_f32_16x16x32_f16 ----
  const int gl = lane & 15;
  union BU { v16h v; _Float16 h[16]; uint4 q[2]; };
  #pragma unroll 1
  for (int nt = 0; nt < NT; ++nt) {
    const int gcol = nt * 16 + gl;
    union CU { v8f v; float f[8]; } acc[STRIPES];
    const float bv = bias[gcol];
    #pragma unroll
    for (int t = 0; t < STRIPES; ++t)
      #pragma unroll
      for (int j = 0; j < 8; ++j) acc[t].f[j] = bv;   // C init = bias

    const float2* wrow = (const float2*)(W + (size_t)gcol * KF);
    const uint4* psrc = (const uint4*)pw + (size_t)(nt * KS * 32 + lane) * 2;

    // B fragment (32x16 f16): lane holds column gcol; K pairs per VGPR,
    // lanes 0-15 -> K=2v, lanes 16-31 -> K=16+2v (within a K-step).
    // Double-buffered so the next load overlaps the two WMMAs.
    BU bf[2];
    if constexpr (PACKED) {
      bf[0].q[0] = psrc[0];
      bf[0].q[1] = psrc[1];
    } else {
      #pragma unroll
      for (int v = 0; v < 8; ++v) {
        float2 wv = wrow[(16 * half + 2 * v) >> 1];
        bf[0].h[2 * v]     = (_Float16)wv.x;
        bf[0].h[2 * v + 1] = (_Float16)wv.y;
      }
    }

    #pragma unroll
    for (int s = 0; s < KS; ++s) {
      if (s + 1 < KS) {  // prefetch next K-step's B fragment
        if constexpr (PACKED) {
          const uint4* nsrc = psrc + (size_t)(s + 1) * 64;  // 32 lanes * 2
          bf[(s + 1) & 1].q[0] = nsrc[0];
          bf[(s + 1) & 1].q[1] = nsrc[1];
        } else {
          #pragma unroll
          for (int v = 0; v < 8; ++v) {
            float2 wv = wrow[(32 * (s + 1) + 16 * half + 2 * v) >> 1];
            bf[(s + 1) & 1].h[2 * v]     = (_Float16)wv.x;
            bf[(s + 1) & 1].h[2 * v + 1] = (_Float16)wv.y;
          }
        }
      }
      #pragma unroll
      for (int t = 0; t < STRIPES; ++t) {
        acc[t].v = __builtin_amdgcn_wmma_f32_16x16x32_f16(
            /*neg_a=*/false, afrag[t][s].v, /*neg_b=*/false, bf[s & 1].v,
            /*c_mod=*/(short)0, acc[t].v, /*reuse_a=*/false, /*reuse_b=*/false);
      }
    }

    // C/D layout: VGPR j -> M = j + 8*(lane/16), N = lane%16
    #pragma unroll
    for (int t = 0; t < STRIPES; ++t) {
      #pragma unroll
      for (int j = 0; j < 8; ++j) {
        out[(size_t)(row0 + t * 16 + j + 8 * half) * NG + gcol] = acc[t].f[j];
      }
    }
  }
}

extern "C" void kernel_launch(void* const* d_in, const int* in_sizes, int n_in,
                              void* d_out, int out_size, void* d_ws, size_t ws_size,
                              hipStream_t stream) {
  const float* x = (const float*)d_in[0];   // [256,512] f32
  const float* W = (const float*)d_in[1];   // [256,256] f32
  const float* b = (const float*)d_in[2];   // [256] f32
  float* out = (float*)d_out;               // [256,512,256] f32

  const int M = in_sizes[0];                // 131072 rows
  const int blocks = M / ROWS_PER_BLOCK;    // 2048

  if (ws_size >= PACKED_BYTES) {
    uint32_t* pw = (uint32_t*)d_ws;
    hipLaunchKernelGGL(pack_W_f16, dim3((int)(PACKED_WORDS / 256)), dim3(256),
                       0, stream, W, pw);
    hipLaunchKernelGGL((float_embed_linear_wmma<true>), dim3(blocks),
                       dim3(WAVES * 32), 0, stream, x, W, pw, b, out);
  } else {
    hipLaunchKernelGGL((float_embed_linear_wmma<false>), dim3(blocks),
                       dim3(WAVES * 32), 0, stream, x, W, nullptr, b, out);
  }
}
